// VectorQuantizer_29119878267084
// MI455X (gfx1250) — matrix operand order repaired and checked by reading, not compiled, not run on hardware
//
#include <hip/hip_runtime.h>
#include <hip/hip_bf16.h>

// VQ-VAE vector quantizer for MI455X (gfx1250).
//   z_e:        [64, 64, 64, 64]  (B, D, H, W) fp32
//   embeddings: [512, 64]         (K, D)       fp32
// Outputs: z_q (== z_q_st forward) [B,D,H,W] fp32, then scalar loss.
//
// Pipeline:
//   vq_prep      : fp32 codebook -> f16 codebook + ||e||^2   (tiny)
//   vq_argmin    : WMMA f16 distance tiles + packed-key argmin -> Idx[N]
//   vq_gather    : z_q = E[idx], per-block partial sum of (z_q - z_e)^2
//   vq_loss_red  : deterministic reduction of partials -> loss scalar

typedef __attribute__((ext_vector_type(16))) _Float16 v16h;
typedef __attribute__((ext_vector_type(8)))  float    v8f;
typedef int i32x4 __attribute__((vector_size(16)));

#define KCODES 512
#define DIM    64     // d dimension
#define EPITCH 72     // padded LDS codebook row pitch in halves (144 B -> 0 bank conflicts)
#define HW     4096   // H*W (stride between d planes, in floats)
#define BSTR   262144 // D*H*W (stride between batches, in floats)

// ---------------- helpers ----------------

// Branchless xor-lane min step via ds_swizzle (group-of-32: and=0x1F, xor=IMM[14:10]).
template <int IMM>
__device__ __forceinline__ float min_step(float v) {
#if __has_builtin(__builtin_amdgcn_ds_swizzle)
    const float o = __int_as_float(__builtin_amdgcn_ds_swizzle(__float_as_int(v), IMM));
#else
    const float o = __shfl_xor(v, (IMM >> 10) & 31, 32);
#endif
    return __builtin_fminf(v, o);
}

// ---------------- Kernel A: codebook prep (1 block, 256 thr) ----------------
__global__ __launch_bounds__(256) void vq_prep(const float* __restrict__ E,
                                               _Float16* __restrict__ Eh,
                                               float* __restrict__ En) {
    const int tid = threadIdx.x;
    for (int i = tid; i < KCODES * DIM; i += 256)
        Eh[i] = (_Float16)E[i];
    for (int r = tid; r < KCODES; r += 256) {
        const float* row = E + r * DIM;
        float s = 0.f;
#pragma unroll
        for (int c = 0; c < DIM; ++c) s = fmaf(row[c], row[c], s);
        En[r] = s;
    }
}

// ---------------- Kernel B: WMMA distances + argmin ----------------
// Grid: 2048 blocks (b in [0,64), h-pair in [0,32)); 256 thr = 8 waves.
// Each wave owns 16 positions; dist_tile = A(-2*z, f16) x B(e, f16) + C(||e||^2).
// Running argmin keeps a single f32 key per row: dist bits with the 9-bit code
// index bitfield-inserted into the low mantissa (v_bfi_b32 + v_min_num_f32).
__global__ __launch_bounds__(256) void vq_argmin(const float* __restrict__ Z,
                                                 const _Float16* __restrict__ Eh,
                                                 const float* __restrict__ En,
                                                 int* __restrict__ Idx) {
    __shared__ _Float16 sE[KCODES * EPITCH];  // 73728 B padded f16 codebook
    __shared__ float    sEn[KCODES];          //  2048 B  ||e||^2

    const int tid = threadIdx.x;
    const int blk = blockIdx.x;
    const int b   = blk >> 5;
    const int h0  = (blk & 31) << 1;

    // Warm caches for this block's z tile (lowers to global_prefetch_b8).
    __builtin_prefetch(Z + (size_t)b * BSTR + (size_t)(tid & 63) * HW
                         + (size_t)h0 * 64, 0, 3);

    // Stage the f16 codebook into LDS (padded rows: 128 B src -> 144 B dst),
    // using the gfx1250 async-to-LDS path (ASYNCcnt) when available.
#if __has_builtin(__builtin_amdgcn_global_load_async_to_lds_b128)
    {
        typedef __attribute__((address_space(1))) i32x4 gvec_t;  // global int4
        typedef __attribute__((address_space(3))) i32x4 lvec_t;  // LDS int4
        gvec_t* gsrc = (gvec_t*)(const void*)Eh;   // 16 B chunks, 8 per src row
        lvec_t* ldst = (lvec_t*)(void*)sE;         // dst row pitch = 9 chunks
        for (int i = tid; i < KCODES * 8; i += 256) {
            const int r = i >> 3, c = i & 7;
            __builtin_amdgcn_global_load_async_to_lds_b128(
                gsrc + i, ldst + r * 9 + c, 0, 0);
        }
#if __has_builtin(__builtin_amdgcn_s_wait_asynccnt)
        __builtin_amdgcn_s_wait_asynccnt(0);
#else
        asm volatile("s_wait_asynccnt 0x0" ::: "memory");
#endif
    }
#else
    {
        const unsigned long long* src = (const unsigned long long*)Eh;
        unsigned long long*       dst = (unsigned long long*)sE;
        for (int i = tid; i < KCODES * 16; i += 256) {       // 8 B chunks
            const int r = i >> 4, c = i & 15;
            dst[r * (EPITCH / 4) + c] = src[i];
        }
    }
#endif
    for (int i = tid; i < KCODES; i += 256) sEn[i] = En[i];
    __syncthreads();

    const int lane = tid & 31;
    const int wave = tid >> 5;
    const int p0   = wave << 4;            // first of this wave's 16 positions
    const int pm   = p0 + (lane & 15);     // A-matrix row owned by this lane
    const int hh   = pm >> 6;
    const int w    = pm & 63;
    const int kb   = (lane < 16) ? 0 : 8;  // ISA 16-bit A-layout K-split
    const size_t zrow = (size_t)b * BSTR + (size_t)(h0 + hh) * 64 + (size_t)w;

    // Build A = f16(-2 * z_row) in the 16x32 f16 A-layout:
    //   lanes 0-15 : K in {0..7, 16..23};  lanes 16-31: K in {8..15, 24..31}
    v16h a_lo, a_hi;
#pragma unroll
    for (int j = 0; j < 8; ++j) {
        a_lo[j]     = (_Float16)(-2.0f * Z[zrow + (size_t)(kb + j)      * HW]);
        a_lo[j + 8] = (_Float16)(-2.0f * Z[zrow + (size_t)(kb + 16 + j) * HW]);
        a_hi[j]     = (_Float16)(-2.0f * Z[zrow + (size_t)(kb + 32 + j) * HW]);
        a_hi[j + 8] = (_Float16)(-2.0f * Z[zrow + (size_t)(kb + 48 + j) * HW]);
    }

    const int ncol = lane & 15;            // this lane's output column (code%16)
    const int kbB  = (lane < 16) ? 0 : 8;  // 32x16 f16 B-layout K-split

    float key[8];                          // packed (dist | code) running min
#pragma unroll
    for (int r = 0; r < 8; ++r) key[r] = __int_as_float(0x7F000000);

#pragma unroll 2
    for (int t = 0; t < 32; ++t) {         // 32 code tiles of 16 codes
        const int n = (t << 4) + ncol;
        const _Float16* ep = sE + n * EPITCH;
        v16h b_lo, b_hi;
#pragma unroll
        for (int j = 0; j < 16; ++j) {
            b_lo[j] = ep[kbB + (j & 7) + ((j >> 3) << 4)];
            b_hi[j] = ep[kbB + 32 + (j & 7) + ((j >> 3) << 4)];
        }
        const float bias = sEn[n];         // ||e_n||^2, same for all 8 acc rows
        v8f c;
#pragma unroll
        for (int r = 0; r < 8; ++r) c[r] = bias;
        // dist - ||z||^2 = ||e||^2 + (-2z).e , f32 accumulate
        c = __builtin_amdgcn_wmma_f32_16x16x32_f16(false, a_lo, false, b_lo,
                                                   (short)0, c, false, false);
        c = __builtin_amdgcn_wmma_f32_16x16x32_f16(false, a_hi, false, b_hi,
                                                   (short)0, c, false, false);
#pragma unroll
        for (int r = 0; r < 8; ++r) {      // v_bfi_b32 + v_min_num_f32
            const float kd = __int_as_float((__float_as_int(c[r]) & ~0x1FF)
                                            | (n & 0x1FF));
            key[r] = __builtin_fminf(key[r], kd);
        }
    }

    // Branchless cross-lane min over the 16-lane column groups (C-layout:
    // VGPR r, lanes 0-15 -> M=r, lanes 16-31 -> M=8+r); decode idx from key.
#pragma unroll
    for (int r = 0; r < 8; ++r) {
        float kv = key[r];
        kv = min_step<0x201F>(kv);         // xor 8
        kv = min_step<0x101F>(kv);         // xor 4
        kv = min_step<0x081F>(kv);         // xor 2
        kv = min_step<0x041F>(kv);         // xor 1
        if ((lane & 15) == 0) {
            const int m = r + ((lane >> 4) << 3);
            const int p = p0 + m;
            Idx[((b << 6) + h0 + (p >> 6)) * 64 + (p & 63)] =
                __float_as_int(kv) & 0x1FF;
        }
    }
}

// ---------------- Kernel C: gather z_q + per-block loss partial ----------------
// Grid: 4096 blocks = (b, h) planes; each block writes 64 d x 64 w elements.
__global__ __launch_bounds__(256) void vq_gather(const float* __restrict__ Z,
                                                 const float* __restrict__ E,
                                                 const int* __restrict__ Idx,
                                                 float* __restrict__ Out,
                                                 float* __restrict__ Part) {
    __shared__ int   sIdx[64];
    __shared__ float sPart[8];
    const int blk = blockIdx.x;            // b*64 + h
    const int b = blk >> 6, h = blk & 63;
    const int tid = threadIdx.x;
    if (tid < 64) sIdx[tid] = Idx[blk * 64 + tid];
    __syncthreads();

    const int w    = tid & 63;
    const int d0   = tid >> 6;             // 0..3
    const int code = sIdx[w];
    const size_t zbase = (size_t)b * BSTR + (size_t)h * 64 + (size_t)w;
    float acc = 0.f;
#pragma unroll
    for (int i = 0; i < 16; ++i) {
        const int    d   = d0 + (i << 2);
        const float  e   = E[code * DIM + d];      // 128 KB table, L2-hot gather
        const size_t off = zbase + (size_t)d * HW;
        const float  z   = __builtin_nontemporal_load(Z + off);  // last use of z
        const float  diff = e - z;
        acc = fmaf(diff, diff, acc);
        __builtin_nontemporal_store(e, Out + off); // write-once stream
    }
    // wave32 reduction, then fixed-order block sum (deterministic)
#pragma unroll
    for (int mask = 16; mask >= 1; mask >>= 1) acc += __shfl_xor(acc, mask, 32);
    if ((tid & 31) == 0) sPart[tid >> 5] = acc;
    __syncthreads();
    if (tid == 0) {
        float s = 0.f;
#pragma unroll
        for (int i = 0; i < 8; ++i) s += sPart[i];
        Part[blk] = s;
    }
}

// ---------------- Kernel D: deterministic final loss reduction ----------------
__global__ __launch_bounds__(256) void vq_loss_reduce(const float* __restrict__ Part,
                                                      float* __restrict__ Loss) {
    __shared__ float s[256];
    float a = 0.f;
    for (int i = threadIdx.x; i < 4096; i += 256) a += Part[i];
    s[threadIdx.x] = a;
    __syncthreads();
    for (int st = 128; st > 0; st >>= 1) {
        if (threadIdx.x < (unsigned)st) s[threadIdx.x] += s[threadIdx.x + st];
        __syncthreads();
    }
    // loss = codebook + 0.25*commit = 1.25 * mean(diff^2) over 64^4 elements
    if (threadIdx.x == 0) *Loss = s[0] * (1.25f / 16777216.0f);
}

extern "C" void kernel_launch(void* const* d_in, const int* in_sizes, int n_in,
                              void* d_out, int out_size, void* d_ws, size_t ws_size,
                              hipStream_t stream) {
    const float* Z = (const float*)d_in[0];   // z_e  [64,64,64,64]
    const float* E = (const float*)d_in[1];   // embeddings [512,64]
    float* Out  = (float*)d_out;              // 64^4 z_q values + 1 loss
    float* Loss = Out + (out_size - 1);

    // Workspace layout (~1.13 MB):
    char*     ws   = (char*)d_ws;
    _Float16* Eh   = (_Float16*)ws;                                   // 65536 B
    float*    En   = (float*)(ws + 65536);                            //  2048 B
    int*      Idx  = (int*)(ws + 65536 + 2048);                       //  1 MB
    float*    Part = (float*)(ws + 65536 + 2048 + 262144 * 4);        // 16 KB

    vq_prep       <<<1,    256, 0, stream>>>(E, Eh, En);
    vq_argmin     <<<2048, 256, 0, stream>>>(Z, Eh, En, Idx);
    vq_gather     <<<4096, 256, 0, stream>>>(Z, E, Idx, Out, Part);
    vq_loss_reduce<<<1,    256, 0, stream>>>(Part, Loss);
}
